// CausalSelfAttention_22806276341923
// MI455X (gfx1250) — matrix operand order repaired and checked by reading, not compiled, not run on hardware
//
#include <hip/hip_runtime.h>
#include <hip/hip_bf16.h>
#include <stdint.h>

typedef __bf16 bf16_t;
typedef __attribute__((ext_vector_type(16))) __bf16 v16bf;
typedef __attribute__((ext_vector_type(8)))  float  v8f;

union FragAB {
  v16bf    v;
  uint32_t u[8];
  uint4    q[2];
};

__device__ __forceinline__ v8f wmma_bf16(v16bf a, v16bf b, v8f c) {
  // D = A(16x32 bf16) * B(32x16 bf16) + C(16x16 f32)
  return __builtin_amdgcn_wmma_f32_16x16x32_bf16(false, a, false, b, (short)0, c,
                                                 false, false);
}

// wave32 xor-shuffle via ds_swizzle (group-of-32: and=0x1f, or=0, xor=MASK)
template<int XMASK>
__device__ __forceinline__ float swz_xor(float x) {
  return __int_as_float(
      __builtin_amdgcn_ds_swizzle(__float_as_int(x), (XMASK << 10) | 0x1f));
}

// ---------------------------------------------------------------- fp32->bf16
__global__ __launch_bounds__(256) void cvt_f32_bf16(const float* __restrict__ in,
                                                    bf16_t* __restrict__ out, int n) {
  int i = blockIdx.x * 256 + threadIdx.x;
  if (i < n) out[i] = (bf16_t)in[i];
}

// ---------------------------------------------------------------- bf16 GEMM
// C(MxN) = A(MxK,row) * B(KxN,row).  Block tile 128x128, K-step 32.
// 8 waves: wave (wr,wc) in 4x2 grid owns a 32x64 sub-tile (2x4 WMMA accs).
// MODE: 0 = store bf16   1 = RoPE + store bf16   2 = store bf16 transposed
//       3 = store f32
constexpr int BM = 128, BN = 128, BK = 32, LPAD = 8;

template<int MODE>
__global__ __launch_bounds__(256)
void gemm_bf16(const bf16_t* __restrict__ A, const bf16_t* __restrict__ Bm,
               bf16_t* __restrict__ outB, float* __restrict__ outF,
               int M, int N, int K,
               const float* __restrict__ fcos, const float* __restrict__ fsin,
               int S) {
  __shared__ bf16_t As[2][BM][BK + LPAD];          // row-major (m,k)
  __shared__ bf16_t Bs[2][BN][BK + LPAD];          // transposed  (n,k)

  const int tid  = threadIdx.x;
  const int lane = tid & 31;
  const int wave = tid >> 5;
  const int wr   = wave >> 1, wc = wave & 1;
  const int bm   = blockIdx.y * BM, bn = blockIdx.x * BN;
  const int l15  = lane & 15, hf = lane >> 4;

  v8f acc[2][4];
#pragma unroll
  for (int a = 0; a < 2; ++a)
#pragma unroll
    for (int b = 0; b < 4; ++b)
#pragma unroll
      for (int e = 0; e < 8; ++e) acc[a][b][e] = 0.0f;

  uint4 aR[2], bR[2];
  const int nk = K / BK;

  auto loadRegs = [&](int kt) {
    const int k0 = kt * BK;
#pragma unroll
    for (int i = 0; i < 2; ++i) {
      int idx = tid + i * 256;
      int r = idx >> 2, cg = idx & 3;                     // A: 128 rows x 4 groups of 8
      aR[i] = *reinterpret_cast<const uint4*>(
          A + (size_t)(bm + r) * K + k0 + cg * 8);
      int kr = idx >> 4, ng = idx & 15;                   // B: 32 k-rows x 16 groups of 8
      bR[i] = *reinterpret_cast<const uint4*>(
          Bm + (size_t)(k0 + kr) * N + bn + ng * 8);
    }
  };
  auto storeLds = [&](int buf) {
#pragma unroll
    for (int i = 0; i < 2; ++i) {
      int idx = tid + i * 256;
      int r = idx >> 2, cg = idx & 3;
      *reinterpret_cast<uint4*>(&As[buf][r][cg * 8]) = aR[i];
      int kr = idx >> 4, ng = idx & 15;
      bf16_t tmp[8];
      *reinterpret_cast<uint4*>(tmp) = bR[i];
#pragma unroll
      for (int j = 0; j < 8; ++j) Bs[buf][ng * 8 + j][kr] = tmp[j];  // transpose
    }
  };
  auto compute = [&](int buf) {
    FragAB af[2], bfg[4];
#pragma unroll
    for (int fr = 0; fr < 2; ++fr) {
      int m = wr * 32 + fr * 16 + l15;
#pragma unroll
      for (int v = 0; v < 8; ++v) {
        int ko = (hf << 4) + (v << 1);  // A-frag: contiguous k = 16*hf + s
        af[fr].u[v] = *reinterpret_cast<const uint32_t*>(&As[buf][m][ko]);
      }
    }
#pragma unroll
    for (int fn = 0; fn < 4; ++fn) {
      int n  = wc * 64 + fn * 16 + l15;
      int kb = hf << 4;                                   // B-frag: contiguous k = 16*hf + s
      bfg[fn].q[0] = *reinterpret_cast<const uint4*>(&Bs[buf][n][kb]);
      bfg[fn].q[1] = *reinterpret_cast<const uint4*>(&Bs[buf][n][kb + 8]);
    }
#pragma unroll
    for (int fr = 0; fr < 2; ++fr)
#pragma unroll
      for (int fn = 0; fn < 4; ++fn)
        acc[fr][fn] = wmma_bf16(af[fr].v, bfg[fn].v, acc[fr][fn]);
  };

  loadRegs(0);
  storeLds(0);
  __syncthreads();
  int buf = 0;
  for (int kt = 0; kt < nk; ++kt) {
    if (kt + 1 < nk) loadRegs(kt + 1);
    compute(buf);
    if (kt + 1 < nk) storeLds(buf ^ 1);
    __syncthreads();
    buf ^= 1;
  }

  // epilogue: C layout -> lane l15 holds col, VGPR r holds row r + hf*8
#pragma unroll
  for (int fr = 0; fr < 2; ++fr) {
#pragma unroll
    for (int fn = 0; fn < 4; ++fn) {
#pragma unroll
      for (int r = 0; r < 8; ++r) {
        int row = bm + wr * 32 + fr * 16 + r + (hf << 3);
        int col = bn + wc * 64 + fn * 16 + l15;
        float v = acc[fr][fn][r];
        if (MODE == 1) {  // fused RoPE (interleaved pairs; partner in lane l^1)
          float p  = swz_xor<1>(v);
          int   s  = row & (S - 1);
          int   d  = col & 63;
          float c  = fcos[s * 32 + (d >> 1)];
          float sn = fsin[s * 32 + (d >> 1)];
          v = (d & 1) ? (p * sn + v * c) : (v * c - p * sn);
          outB[(size_t)row * N + col] = (bf16_t)v;
        } else if (MODE == 2) {  // V stored transposed: [b][dim][s]
          int bb = row / S, srow = row - bb * S;
          outB[((size_t)bb * N + col) * S + srow] = (bf16_t)v;
        } else if (MODE == 3) {
          outF[(size_t)row * N + col] = v;
        } else {
          outB[(size_t)row * N + col] = (bf16_t)v;
        }
      }
    }
  }
}

// ---------------------------------------------------------------- flash attn
// grid(S/64, H=32, B=2), block = 128 threads = 4 waves; wave owns 16 queries.
// Q: (B*S, 2048) bf16 rope'd.  K: (B*S, 512) bf16 rope'd.  Vt: [b][512][S] bf16.
__global__ __launch_bounds__(128)
void flash_attn(const bf16_t* __restrict__ Q, const bf16_t* __restrict__ Kb,
                const bf16_t* __restrict__ Vt, bf16_t* __restrict__ Ao, int S) {
  __shared__ bf16_t pbuf[4][16][40];  // per-wave C->A relayout scratch

  const int lane  = threadIdx.x & 31, w = threadIdx.x >> 5;
  const int l15   = lane & 15, hf = lane >> 4;
  const int qblk  = blockIdx.x, h = blockIdx.y, bIdx = blockIdx.z;
  const int qBase = qblk * 64 + w * 16;
  const int kvCol = (h >> 2) * 64;  // GQA: 4 query heads per KV head
  const float SCL = 0.125f * 1.44269504088896340736f;  // 1/sqrt(64) * log2(e)

  // Q A-fragments for head-dim halves k=[0,32) and [32,64); resident all loop
  FragAB qf[2];
#pragma unroll
  for (int kf = 0; kf < 2; ++kf)
#pragma unroll
    for (int v = 0; v < 8; ++v) {
      int ko = kf * 32 + (hf << 4) + (v << 1);
      qf[kf].u[v] = *reinterpret_cast<const uint32_t*>(
          Q + (size_t)(bIdx * S + qBase + l15) * 2048 + h * 64 + ko);
    }

  v8f o[4];
#pragma unroll
  for (int t = 0; t < 4; ++t)
#pragma unroll
    for (int e = 0; e < 8; ++e) o[t][e] = 0.0f;
  float mrow[8], lrow[8];
#pragma unroll
  for (int r = 0; r < 8; ++r) { mrow[r] = -1.0e30f; lrow[r] = 0.0f; }

  const int nkb = (qBase + 47) >> 5;  // ceil((qBase+16)/32) key blocks
  for (int jb = 0; jb < nkb; ++jb) {
    const int j0 = jb << 5;

    // K^T B-fragments: 2 key sub-tiles x 2 head-dim halves
    FragAB kfrag[2][2];
#pragma unroll
    for (int nh = 0; nh < 2; ++nh) {
      int key = j0 + nh * 16 + l15;
#pragma unroll
      for (int kf = 0; kf < 2; ++kf) {
        const bf16_t* p =
            Kb + (size_t)(bIdx * S + key) * 512 + kvCol + kf * 32 + (hf << 4);
        kfrag[nh][kf].q[0] = *reinterpret_cast<const uint4*>(p);
        kfrag[nh][kf].q[1] = *reinterpret_cast<const uint4*>(p + 8);
      }
    }

    v8f s0, s1;
#pragma unroll
    for (int e = 0; e < 8; ++e) { s0[e] = 0.0f; s1[e] = 0.0f; }
    s0 = wmma_bf16(qf[0].v, kfrag[0][0].v, s0);
    s0 = wmma_bf16(qf[1].v, kfrag[0][1].v, s0);
    s1 = wmma_bf16(qf[0].v, kfrag[1][0].v, s1);
    s1 = wmma_bf16(qf[1].v, kfrag[1][1].v, s1);

    if (j0 + 31 > qBase) {  // causal mask on diagonal blocks
#pragma unroll
      for (int r = 0; r < 8; ++r) {
        int qrow = qBase + r + (hf << 3);
        int k0e  = j0 + l15;
        if (k0e > qrow)      s0[r] = -3.0e38f;
        if (k0e + 16 > qrow) s1[r] = -3.0e38f;
      }
    }

    // online softmax (row = r + hf*8; reduce across the 16 lanes of a half)
    float alpha[8];
#pragma unroll
    for (int r = 0; r < 8; ++r) {
      float v = fmaxf(s0[r], s1[r]);
      v = fmaxf(v, swz_xor<1>(v));
      v = fmaxf(v, swz_xor<2>(v));
      v = fmaxf(v, swz_xor<4>(v));
      v = fmaxf(v, swz_xor<8>(v));
      float mn = fmaxf(mrow[r], v);
      alpha[r] = exp2f((mrow[r] - mn) * SCL);
      mrow[r]  = mn;
      s0[r] = exp2f((s0[r] - mn) * SCL);
      s1[r] = exp2f((s1[r] - mn) * SCL);
      float ps = s0[r] + s1[r];
      ps += swz_xor<1>(ps);
      ps += swz_xor<2>(ps);
      ps += swz_xor<4>(ps);
      ps += swz_xor<8>(ps);
      lrow[r] = lrow[r] * alpha[r] + ps;
    }
#pragma unroll
    for (int t = 0; t < 4; ++t)
#pragma unroll
      for (int r = 0; r < 8; ++r) o[t][r] *= alpha[r];

    // P (C-layout f32) -> bf16 A-fragment via per-wave LDS tile
#pragma unroll
    for (int r = 0; r < 8; ++r) {
      int row = r + (hf << 3);
      pbuf[w][row][l15]      = (bf16_t)s0[r];
      pbuf[w][row][16 + l15] = (bf16_t)s1[r];
    }
    asm volatile("s_wait_dscnt 0" ::: "memory");  // wave-local store->load
    FragAB pf;
#pragma unroll
    for (int v = 0; v < 8; ++v) {
      int ko = (hf << 4) + (v << 1);
      pf.u[v] = *reinterpret_cast<const uint32_t*>(&pbuf[w][l15][ko]);
    }

    // P(16x32) x V(32x64): V B-frags contiguous thanks to transposed Vt
#pragma unroll
    for (int t = 0; t < 4; ++t) {
      FragAB vf;
      const bf16_t* p =
          Vt + ((size_t)bIdx * 512 + kvCol + t * 16 + l15) * S + j0 + (hf << 4);
      vf.q[0] = *reinterpret_cast<const uint4*>(p);
      vf.q[1] = *reinterpret_cast<const uint4*>(p + 8);
      o[t] = wmma_bf16(pf.v, vf.v, o[t]);
    }
  }

  float inv[8];
#pragma unroll
  for (int r = 0; r < 8; ++r) inv[r] = 1.0f / lrow[r];
#pragma unroll
  for (int t = 0; t < 4; ++t)
#pragma unroll
    for (int r = 0; r < 8; ++r)
      Ao[(size_t)(bIdx * S + qBase + r + (hf << 3)) * 2048 + h * 64 + t * 16 +
         l15] = (bf16_t)(o[t][r] * inv[r]);
}

// ---------------------------------------------------------------- launcher
extern "C" void kernel_launch(void* const* d_in, const int* in_sizes, int n_in,
                              void* d_out, int out_size, void* d_ws,
                              size_t ws_size, hipStream_t stream) {
  (void)in_sizes; (void)n_in; (void)out_size; (void)ws_size;
  const float* x    = (const float*)d_in[0];
  const float* wq   = (const float*)d_in[1];
  const float* wk   = (const float*)d_in[2];
  const float* wv   = (const float*)d_in[3];
  const float* wo   = (const float*)d_in[4];
  const float* fcos = (const float*)d_in[5];
  const float* fsin = (const float*)d_in[6];

  const int Bt = 2, S = 2048, E = 2048, KV = 512;
  const int M = Bt * S;  // 4096

  char*  ws  = (char*)d_ws;
  size_t off = 0;
  auto alloc = [&](size_t bytes) {
    void* p = ws + off;
    off += (bytes + 255) & ~(size_t)255;
    return p;
  };
  bf16_t* xb   = (bf16_t*)alloc((size_t)M * E * 2);
  bf16_t* wqb  = (bf16_t*)alloc((size_t)E * E * 2);
  bf16_t* wkb  = (bf16_t*)alloc((size_t)E * KV * 2);
  bf16_t* wvb  = (bf16_t*)alloc((size_t)E * KV * 2);
  bf16_t* wob  = (bf16_t*)alloc((size_t)E * E * 2);
  bf16_t* Qb   = (bf16_t*)alloc((size_t)M * E * 2);
  bf16_t* Kbuf = (bf16_t*)alloc((size_t)M * KV * 2);
  bf16_t* Vt   = (bf16_t*)alloc((size_t)M * KV * 2);
  bf16_t* attn = (bf16_t*)alloc((size_t)M * E * 2);

  auto cvt = [&](const float* src, bf16_t* dst, int n) {
    cvt_f32_bf16<<<(n + 255) / 256, 256, 0, stream>>>(src, dst, n);
  };
  cvt(x, xb, M * E);
  cvt(wq, wqb, E * E);
  cvt(wk, wkb, E * KV);
  cvt(wv, wvb, E * KV);
  cvt(wo, wob, E * E);

  dim3 blk(256);
  // Q = x@wq with fused RoPE
  gemm_bf16<1><<<dim3(E / BN, M / BM), blk, 0, stream>>>(
      xb, wqb, Qb, nullptr, M, E, E, fcos, fsin, S);
  // K = x@wk with fused RoPE
  gemm_bf16<1><<<dim3(KV / BN, M / BM), blk, 0, stream>>>(
      xb, wkb, Kbuf, nullptr, M, KV, E, fcos, fsin, S);
  // V = x@wv, stored transposed [b][dim][s]
  gemm_bf16<2><<<dim3(KV / BN, M / BM), blk, 0, stream>>>(
      xb, wvb, Vt, nullptr, M, KV, E, nullptr, nullptr, S);
  // flash attention
  flash_attn<<<dim3(S / 64, 32, Bt), dim3(128), 0, stream>>>(Qb, Kbuf, Vt, attn, S);
  // out = attn @ wo -> fp32
  gemm_bf16<3><<<dim3(E / BN, M / BM), blk, 0, stream>>>(
      attn, wob, nullptr, (float*)d_out, M, E, E, nullptr, nullptr, S);
}
